// FeedbackTransformerKV_66322884985019
// MI455X (gfx1250) — matrix operand order repaired and checked by reading, not compiled, not run on hardware
//
#include <hip/hip_runtime.h>
#include <hip/hip_bf16.h>
#include <cstdint>

// Problem constants (match reference)
#define S_LEN 128
#define BSZ   64
#define DMODEL 512
#define NHEAD 8
#define NLAYER 2
#define FFN   2048
#define PMAX  4096
#define DK    64

typedef __bf16 bf16_t;
typedef __attribute__((ext_vector_type(16))) __bf16 v16bf;
typedef __attribute__((ext_vector_type(8)))  float  v8f;
typedef __attribute__((ext_vector_type(4)))  unsigned int u32x4;
typedef __attribute__((ext_vector_type(8)))  int  i32x8;
typedef __attribute__((ext_vector_type(4)))  int  i32x4;

// ---- Tensor Data Mover availability probes (toolchain-dependent) ----
#if defined(__has_builtin)
#  if __has_builtin(__builtin_amdgcn_tensor_load_to_lds)
#    define FBT_HAS_TDM 1
#  else
#    define FBT_HAS_TDM 0
#  endif
#else
#  define FBT_HAS_TDM 0
#endif

#if __has_include(<hip/amd_detail/amd_gfx1250_TDM.h>)
#  define FBT_TDM_6ARG 1   // therock/clang-23 headers present -> 6-arg builtin
#else
#  define FBT_TDM_6ARG 0   // ROCm 7.2 / clang-22 -> 5-arg builtin
#endif

// ---------------------------------------------------------------------------
// f32 -> bf16 weight conversion
// ---------------------------------------------------------------------------
__global__ void fbt_f2bf_kernel(const float* __restrict__ in, bf16_t* __restrict__ out, int n) {
    int i = blockIdx.x * 256 + threadIdx.x;
    if (i < n) out[i] = (bf16_t)in[i];
}

// softmax over layer_weights (L+1 == 3 entries)
__global__ void fbt_lw_kernel(const float* __restrict__ lwin, float* __restrict__ lw) {
    float m = fmaxf(fmaxf(lwin[0], lwin[1]), lwin[2]);
    float e0 = __expf(lwin[0] - m), e1 = __expf(lwin[1] - m), e2 = __expf(lwin[2] - m);
    float s = e0 + e1 + e2;
    lw[0] = e0 / s; lw[1] = e1 / s; lw[2] = e2 / s;
}

__global__ void fbt_step_init_kernel(const float* __restrict__ xs_t, float* __restrict__ x,
                                     float* __restrict__ macc, const float* __restrict__ lw) {
    int i = blockIdx.x * 256 + threadIdx.x;   // n = BSZ*DMODEL = 32768
    float v = xs_t[i];
    x[i] = v;
    macc[i] = lw[0] * v;
}

__global__ void fbt_axpy_lw_kernel(float* __restrict__ macc, const float* __restrict__ x,
                                   const float* __restrict__ lw, int idx) {
    int i = blockIdx.x * 256 + threadIdx.x;
    macc[i] += lw[idx] * x[i];
}

__global__ void fbt_copy_kernel(const float* __restrict__ src, float* __restrict__ dst) {
    int i = blockIdx.x * 256 + threadIdx.x;
    dst[i] = src[i];
}

// ---------------------------------------------------------------------------
// LayerNorm: one block (256 threads) per row of 512 elements
// ---------------------------------------------------------------------------
__global__ __launch_bounds__(256) void fbt_ln_kernel(const float* __restrict__ x,
                                                     const float* __restrict__ g,
                                                     const float* __restrict__ bta,
                                                     float* __restrict__ z) {
    const int row = blockIdx.x;
    const int tid = threadIdx.x;
    const float* xr = x + (size_t)row * DMODEL;
    __shared__ float rs[256];
    __shared__ float rq[256];
    float a0 = xr[tid], a1 = xr[tid + 256];
    rs[tid] = a0 + a1;
    rq[tid] = a0 * a0 + a1 * a1;
    __syncthreads();
    for (int o = 128; o > 0; o >>= 1) {
        if (tid < o) { rs[tid] += rs[tid + o]; rq[tid] += rq[tid + o]; }
        __syncthreads();
    }
    float mean = rs[0] * (1.0f / DMODEL);
    float var  = rq[0] * (1.0f / DMODEL) - mean * mean;
    float inv  = __frsqrt_rn(var + 1e-5f);
    float* zr = z + (size_t)row * DMODEL;
    zr[tid]       = (a0 - mean) * inv * g[tid]       + bta[tid];
    zr[tid + 256] = (a1 - mean) * inv * g[tid + 256] + bta[tid + 256];
}

// ---------------------------------------------------------------------------
// WMMA GEMM: C[64,N] (+)= relu?( A[64,K](f32->bf16) x Bw[K,N](bf16) + bias )
// Block = 8 waves, covers 64 rows x 64 cols of C. Each wave owns a 16x32
// strip (two 16x16 tiles, one shared A fragment -> two v_wmma per K-chunk).
// B K-chunks (32x64 bf16, LDS row stride 66 = TDM pad of 1 dword / 32 dwords)
// are DMA'd by the Tensor Data Mover with software double buffering:
// chunk i+1 is issued before computing chunk i; s_wait_tensorcnt<=1 leaves
// the new DMA in flight (TDM ops from one wave complete in order).
// ---------------------------------------------------------------------------
template<bool RELU, bool ACCUM, bool BIAS>
__global__ __launch_bounds__(256) void fbt_gemm_bf16_kernel(
    const float*  __restrict__ A,    // [64, K] row-major f32
    const bf16_t* __restrict__ Bw,   // [K, N]  row-major bf16
    const float*  __restrict__ bias, // [N] (used iff BIAS)
    float*        __restrict__ C,    // [64, N]
    int N, int K)
{
    const int lane   = threadIdx.x & 31;
    const int wave   = threadIdx.x >> 5;      // 0..7
    const int tid    = threadIdx.x;           // 0..255
    const int mt     = wave & 3;              // M tile 0..3 (M = 64)
    const int ntPair = wave >> 2;             // 0 or 1 -> 32-col half
    const int nBase  = blockIdx.x * 64;       // block covers 64 columns

    __shared__ bf16_t sA[64 * 34];            // 64 rows x 32 k, pad to 34
    __shared__ bf16_t sB[2][32 * 66];         // ping-pong 32 k x 64 n (TDM pad)

#if FBT_HAS_TDM
    // Issue one TDM descriptor: B tile [k0..k0+32) x [nBase..nBase+64) -> sB[buf]
    auto tdm_issue = [&](int k0, int buf) {
        unsigned long long gaddr = (unsigned long long)(uintptr_t)Bw
            + 2ull * ((unsigned long long)(unsigned)k0 * (unsigned)N + (unsigned)nBase);
        unsigned ldsOff = (unsigned)(uintptr_t)(&sB[buf][0]);
        u32x4 g0;
        g0[0] = 1u;                                   // count=1 (valid D#)
        g0[1] = ldsOff;                               // lds_addr
        g0[2] = (unsigned)gaddr;                      // global_addr[31:0]
        g0[3] = (unsigned)((gaddr >> 32) & 0x1FFFFFFu) | (2u << 30); // [56:32] | type=2
        i32x8 g1;
        g1[0] = (int)((1u << 16) | (1u << 20) | (4u << 22)); // data_size=2B, pad_en, pad_interval=32dw (pad_amount=1dw)
        g1[1] = (int)(((unsigned)N & 0xFFFFu) << 16);        // tensor_dim0[15:0]
        g1[2] = (int)(((unsigned)N >> 16) | (((unsigned)K & 0xFFFFu) << 16)); // td0 hi | td1 lo
        g1[3] = (int)(((unsigned)K >> 16) | (64u << 16));    // td1 hi | tile_dim0=64
        g1[4] = 32;                                          // tile_dim1=32, tile_dim2=0
        g1[5] = (int)(unsigned)N;                            // tensor_dim0_stride[31:0]
        g1[6] = 0;                                           // stride hi / dim1_stride lo
        g1[7] = 0;
        i32x4 gz = {0, 0, 0, 0};
#if FBT_TDM_6ARG
        i32x8 gz8 = {0, 0, 0, 0, 0, 0, 0, 0};
        __builtin_amdgcn_tensor_load_to_lds(g0, g1, gz, gz, gz8, 0);
#else
        __builtin_amdgcn_tensor_load_to_lds(g0, g1, gz, gz, 0);
#endif
    };
#endif

    v8f acc0 = {};
    v8f acc1 = {};

    const int nch = K >> 5;                   // number of 32-wide K chunks

#if FBT_HAS_TDM
    if (wave == 0) tdm_issue(0, 0);           // prefill pipeline
#endif

    for (int ic = 0; ic < nch; ++ic) {
        const int k0  = ic << 5;
        const int buf = ic & 1;

        // cooperative load + convert A chunk: 64x32 f32 -> bf16
        #pragma unroll
        for (int i = 0; i < 8; ++i) {
            int idx = tid * 8 + i;            // 0..2047
            int r = idx >> 5, kk = idx & 31;
            sA[r * 34 + kk] = (bf16_t)A[(size_t)r * K + k0 + kk];
        }
        if (k0 + 32 < K)                      // prefetch next A chunk
            __builtin_prefetch(&A[(size_t)(tid >> 2) * K + k0 + 32], 0, 1);

#if FBT_HAS_TDM
        if (wave == 0) {
            if (ic + 1 < nch) {
                tdm_issue(k0 + 32, buf ^ 1);              // overlap next DMA
                __builtin_amdgcn_s_wait_tensorcnt(1);     // chunk ic done, ic+1 in flight
            } else {
                __builtin_amdgcn_s_wait_tensorcnt(0);     // drain last chunk
            }
        }
#else
        // fallback: cooperative vector load of B chunk 32(K) x 64(N)
        #pragma unroll
        for (int i = 0; i < 8; ++i) {
            int idx = tid * 8 + i;            // 0..2047
            int kk = idx >> 6, cc = idx & 63;
            sB[buf][kk * 66 + cc] = Bw[(size_t)(k0 + kk) * N + nBase + cc];
        }
        if (k0 + 32 < K)
            __builtin_prefetch(&Bw[(size_t)(k0 + 32 + (tid >> 3)) * N + nBase], 0, 1);
#endif
        __syncthreads();

        // A fragment: contiguous k order (k = 16*(lane/16) + e), matching B
        v16bf af;
        const int m  = mt * 16 + (lane & 15);
        const int kb = (lane >> 4) * 16;
        #pragma unroll
        for (int e = 0; e < 8; ++e) {
            af[e]     = sA[m * 34 + kb + e];
            af[e + 8] = sA[m * 34 + kb + 8 + e];
        }
        // B fragments for the two 16-wide column tiles
        v16bf b0, b1;
        const int kr  = (lane >> 4) * 16;
        const int nl0 = ntPair * 32 + (lane & 15);
        #pragma unroll
        for (int e = 0; e < 16; ++e) {
            b0[e] = sB[buf][(kr + e) * 66 + nl0];
            b1[e] = sB[buf][(kr + e) * 66 + nl0 + 16];
        }

        acc0 = __builtin_amdgcn_wmma_f32_16x16x32_bf16(false, af, false, b0,
                                                       (short)0, acc0, false, false);
        acc1 = __builtin_amdgcn_wmma_f32_16x16x32_bf16(false, af, false, b1,
                                                       (short)0, acc1, false, false);
        __syncthreads();
    }

    // Epilogue per C/D layout: row = mt*16 + (lane>=16 ? 8 : 0) + r, col = lane&15
    const int rowBase = mt * 16 + (lane >> 4) * 8;
    const int col0    = nBase + ntPair * 32 + (lane & 15);
    const float bv0   = BIAS ? bias[col0]      : 0.0f;
    const float bv1   = BIAS ? bias[col0 + 16] : 0.0f;
    #pragma unroll
    for (int r = 0; r < 8; ++r) {
        float v0 = acc0[r] + bv0;
        float v1 = acc1[r] + bv1;
        if (RELU) { v0 = fmaxf(v0, 0.0f); v1 = fmaxf(v1, 0.0f); }
        size_t o = (size_t)(rowBase + r) * N + col0;
        if (ACCUM) { C[o] += v0; C[o + 16] += v1; }
        else       { C[o]  = v0; C[o + 16]  = v1; }
    }
}

// ---------------------------------------------------------------------------
// Feedback attention over KV memory for one step t (t > 0).
// grid = B*H blocks, 128 threads. Thread j computes score_j; softmax over j<t;
// then threads 0..63 compute av[d].
// ---------------------------------------------------------------------------
__global__ __launch_bounds__(128) void fbt_attn_kernel(
    const float* __restrict__ q,    // [B, D]
    const float* __restrict__ qpb,  // [H, DK]  (layer slice)
    const float* __restrict__ kmem, // [S, B, H, DK]
    const float* __restrict__ vmem, // [S, B, H, DK]
    const float* __restrict__ kpe,  // [P, H, DK] (layer slice)
    const float* __restrict__ kpb,  // [P, H]     (layer slice)
    float* __restrict__ av,         // [B, D]
    int t)
{
    const int b = blockIdx.x >> 3;
    const int h = blockIdx.x & 7;
    const int tid = threadIdx.x;

    __shared__ float qs[DK];   // q + qpb
    __shared__ float qr[DK];   // raw q
    __shared__ float sc[S_LEN];
    __shared__ float red[128];

    if (tid < DK) {
        float qv = q[(size_t)b * DMODEL + h * DK + tid];
        qr[tid] = qv;
        qs[tid] = qv + qpb[h * DK + tid];
    }
    __syncthreads();

    float s = -3.0e38f;
    if (tid < t) {
        const float* kv = kmem + ((size_t)tid * BSZ + b) * DMODEL + h * DK;
        const int    p  = PMAX - t + tid;
        const float* ke = kpe + ((size_t)p * NHEAD + h) * DK;
        float ac = 0.f, bd = 0.f;
        #pragma unroll
        for (int d = 0; d < DK; ++d) {
            ac += qs[d] * kv[d];
            bd += qr[d] * ke[d];
        }
        s = (ac + bd + kpb[p * NHEAD + h]) * 0.125f;   // 1/sqrt(64)
    }
    red[tid] = s;
    __syncthreads();
    for (int o = 64; o > 0; o >>= 1) {
        if (tid < o) red[tid] = fmaxf(red[tid], red[tid + o]);
        __syncthreads();
    }
    const float mx = red[0];
    __syncthreads();

    const float e = (tid < t) ? __expf(s - mx) : 0.f;
    sc[tid]  = e;
    red[tid] = e;
    __syncthreads();
    for (int o = 64; o > 0; o >>= 1) {
        if (tid < o) red[tid] += red[tid + o];
        __syncthreads();
    }
    const float denom = red[0];
    __syncthreads();

    if (tid < DK) {
        float sum = 0.f;
        for (int j = 0; j < t; ++j)
            sum += sc[j] * vmem[((size_t)j * BSZ + b) * DMODEL + h * DK + tid];
        av[(size_t)b * DMODEL + h * DK + tid] = sum / denom;
    }
}

// ---------------------------------------------------------------------------
// Host-side orchestration
// ---------------------------------------------------------------------------
extern "C" void kernel_launch(void* const* d_in, const int* in_sizes, int n_in,
                              void* d_out, int out_size, void* d_ws, size_t ws_size,
                              hipStream_t stream) {
    (void)in_sizes; (void)n_in; (void)out_size; (void)ws_size;

    const float* x_seq     = (const float*)d_in[0];
    const float* ln_attn_g = (const float*)d_in[1];
    const float* ln_attn_b = (const float*)d_in[2];
    const float* wq        = (const float*)d_in[3];
    const float* w_out     = (const float*)d_in[4];
    const float* b_out     = (const float*)d_in[5];
    const float* kpe       = (const float*)d_in[6];
    const float* kpb       = (const float*)d_in[7];
    const float* qpb       = (const float*)d_in[8];
    const float* ln_ff_g   = (const float*)d_in[9];
    const float* ln_ff_b   = (const float*)d_in[10];
    const float* w1        = (const float*)d_in[11];
    const float* b1        = (const float*)d_in[12];
    const float* w2        = (const float*)d_in[13];
    const float* b2        = (const float*)d_in[14];
    const float* layer_w   = (const float*)d_in[15];
    const float* w_key     = (const float*)d_in[16];
    const float* w_value   = (const float*)d_in[17];
    const float* ln_f_g    = (const float*)d_in[18];
    const float* ln_f_b    = (const float*)d_in[19];

    // ---- carve workspace ----
    char* ws = (char*)d_ws;
    size_t off = 0;
    auto carve = [&](size_t bytes) -> void* {
        void* p = ws + off;
        off = (off + bytes + 255) & ~(size_t)255;
        return p;
    };
    const size_t nWq  = (size_t)NLAYER * DMODEL * DMODEL;
    const size_t nW1  = (size_t)NLAYER * DMODEL * FFN;
    const size_t nWkv = (size_t)DMODEL * DMODEL;
    const size_t nBD  = (size_t)BSZ * DMODEL;          // 32768
    const size_t nSBD = (size_t)S_LEN * BSZ * DMODEL;  // 4M

    bf16_t* wq_bf   = (bf16_t*)carve(nWq  * sizeof(bf16_t));
    bf16_t* wout_bf = (bf16_t*)carve(nWq  * sizeof(bf16_t));
    bf16_t* w1_bf   = (bf16_t*)carve(nW1  * sizeof(bf16_t));
    bf16_t* w2_bf   = (bf16_t*)carve(nW1  * sizeof(bf16_t));
    bf16_t* wkey_bf = (bf16_t*)carve(nWkv * sizeof(bf16_t));
    bf16_t* wval_bf = (bf16_t*)carve(nWkv * sizeof(bf16_t));

    float* kmem = (float*)carve(nSBD * sizeof(float));
    float* vmem = (float*)carve(nSBD * sizeof(float));
    float* ys   = (float*)carve(nSBD * sizeof(float));
    float* x    = (float*)carve(nBD * sizeof(float));
    float* z    = (float*)carve(nBD * sizeof(float));
    float* q    = (float*)carve(nBD * sizeof(float));
    float* av   = (float*)carve(nBD * sizeof(float));
    float* macc = (float*)carve(nBD * sizeof(float));
    float* h1   = (float*)carve((size_t)BSZ * FFN * sizeof(float));
    float* lw   = (float*)carve(16 * sizeof(float));

    // ---- weight conversion (once per launch; deterministic) ----
    auto conv = [&](const float* src, bf16_t* dst, size_t n) {
        fbt_f2bf_kernel<<<(unsigned)((n + 255) / 256), 256, 0, stream>>>(src, dst, (int)n);
    };
    conv(wq,      wq_bf,   nWq);
    conv(w_out,   wout_bf, nWq);
    conv(w1,      w1_bf,   nW1);
    conv(w2,      w2_bf,   nW1);
    conv(w_key,   wkey_bf, nWkv);
    conv(w_value, wval_bf, nWkv);
    fbt_lw_kernel<<<1, 1, 0, stream>>>(layer_w, lw);

    auto gemm = [&](const float* Am, const bf16_t* Bm, const float* bias, float* Cm,
                    int N, int K, bool relu, bool accum) {
        unsigned g = (unsigned)(N / 64);
        if (relu)       fbt_gemm_bf16_kernel<true,  false, true ><<<g, 256, 0, stream>>>(Am, Bm, bias, Cm, N, K);
        else if (accum) fbt_gemm_bf16_kernel<false, true,  true ><<<g, 256, 0, stream>>>(Am, Bm, bias, Cm, N, K);
        else            fbt_gemm_bf16_kernel<false, false, false><<<g, 256, 0, stream>>>(Am, Bm, nullptr, Cm, N, K);
    };

    const unsigned gBD = (unsigned)(nBD / 256);       // 128 blocks of 256

    // ---- sequential scan over time ----
    for (int t = 0; t < S_LEN; ++t) {
        fbt_step_init_kernel<<<gBD, 256, 0, stream>>>(x_seq + (size_t)t * nBD, x, macc, lw);

        for (int l = 0; l < NLAYER; ++l) {
            if (t > 0) {
                fbt_ln_kernel<<<BSZ, 256, 0, stream>>>(x, ln_attn_g + (size_t)l * DMODEL,
                                                       ln_attn_b + (size_t)l * DMODEL, z);
                gemm(z, wq_bf + (size_t)l * DMODEL * DMODEL, nullptr, q, DMODEL, DMODEL, false, false);
                fbt_attn_kernel<<<BSZ * NHEAD, 128, 0, stream>>>(
                    q, qpb + (size_t)l * NHEAD * DK, kmem, vmem,
                    kpe + (size_t)l * PMAX * NHEAD * DK, kpb + (size_t)l * PMAX * NHEAD, av, t);
                gemm(av, wout_bf + (size_t)l * DMODEL * DMODEL, b_out + (size_t)l * DMODEL,
                     x, DMODEL, DMODEL, false, true /*accum*/);
            }
            fbt_ln_kernel<<<BSZ, 256, 0, stream>>>(x, ln_ff_g + (size_t)l * DMODEL,
                                                   ln_ff_b + (size_t)l * DMODEL, z);
            gemm(z, w1_bf + (size_t)l * DMODEL * FFN, b1 + (size_t)l * FFN,
                 h1, FFN, DMODEL, true /*relu*/, false);
            gemm(h1, w2_bf + (size_t)l * FFN * DMODEL, b2 + (size_t)l * DMODEL,
                 x, DMODEL, FFN, false, true /*accum*/);
            fbt_axpy_lw_kernel<<<gBD, 256, 0, stream>>>(macc, x, lw, l + 1);
        }

        fbt_copy_kernel<<<gBD, 256, 0, stream>>>(x, ys + (size_t)t * nBD);
        // KV projections of layer-mixed state into memory slot t
        gemm(macc, wkey_bf, nullptr, kmem + (size_t)t * nBD, DMODEL, DMODEL, false, false);
        gemm(macc, wval_bf, nullptr, vmem + (size_t)t * nBD, DMODEL, DMODEL, false, false);
    }

    // final layer norm over [S*B, D] rows -> d_out
    fbt_ln_kernel<<<S_LEN * BSZ, 256, 0, stream>>>(ys, ln_f_g, ln_f_b, (float*)d_out);
}
